// SPDetector_14293651161642
// MI455X (gfx1250) — matrix operand order repaired and checked by reading, not compiled, not run on hardware
//
#include <hip/hip_runtime.h>
#include <hip/hip_bf16.h>
#include <math.h>

// ---------------- problem constants ----------------
#define BATCH 16
#define ICH   128
#define MIDC  256
#define HH    48
#define WW    64
#define PH    50          // padded H
#define PW    66          // padded W
#define KTOT  1152        // 128 * 9
#define NTOT  (BATCH*HH*WW)   // 49152 spatial positions
#define NCLS  65
#define MPAD  80          // 65 padded to 5 WMMA tiles
#define HMH   384
#define HMW   512
#define HMN   (HMH*HMW)   // 196608
#define SORTN 262144      // 2^18 >= HMN
#define CONF_TH 0.015f

// ---------------- workspace layout (bytes, 256-aligned) ----------------
#define XB_OFF   0ull              // bf16 padded input  16*128*50*66
#define W1B_OFF  13516800ull       // bf16 w1 [256][1152]
#define W2B_OFF  14106624ull       // bf16 w2 [80][256]
#define Y_OFF    14147584ull       // bf16 conv1 out, layout [N=49152][K=256]
#define HM_OFF   39313408ull       // f32 heatmap 16*384*512
#define KEY_OFF  51896320ull       // f32 sort keys 16*2^18
#define VAL_OFF  68673536ull       // i32 sort vals 16*2^18
#define KEEP_OFF 85450752ull       // u8  keep map 16*196608

typedef __bf16 v16bf __attribute__((ext_vector_type(16)));
typedef float  v8f   __attribute__((ext_vector_type(8)));

__device__ __forceinline__ __bf16 f2bf(float f) {
  union { float f; unsigned u; } v; v.f = f;
  unsigned r = v.u + 0x7FFFu + ((v.u >> 16) & 1u);   // RNE
  unsigned short h = (unsigned short)(r >> 16);
  __bf16 o; __builtin_memcpy(&o, &h, 2);
  return o;
}

__device__ __forceinline__ v16bf pack16(uint4 a0, uint4 a1) {
  v16bf v;
  __builtin_memcpy(&v, &a0, 16);
  __builtin_memcpy((char*)&v + 16, &a1, 16);
  return v;
}

// ---------------- prep: pad + convert x to bf16 ----------------
__global__ void k_prep_x(const float* __restrict__ x, __bf16* __restrict__ xb) {
  int idx = blockIdx.x * 256 + threadIdx.x;
  if (idx >= BATCH * ICH * PH * PW) return;
  int pw = idx % PW; int t = idx / PW;
  int ph = t % PH;   t /= PH;
  int c  = t % ICH;  int b = t / ICH;
  int h = ph - 1, w = pw - 1;
  float v = 0.f;
  if (h >= 0 && h < HH && w >= 0 && w < WW)
    v = x[((size_t)(b * ICH + c) * HH + h) * WW + w];
  xb[idx] = f2bf(v);
}

// ---------------- prep: repack weights to bf16 GEMM layout ----------------
__global__ void k_prep_w(const float* __restrict__ w1, const float* __restrict__ w2,
                         __bf16* __restrict__ w1b, __bf16* __restrict__ w2b) {
  int idx = blockIdx.x * 256 + threadIdx.x;
  if (idx < MIDC * KTOT) {          // w1 -> [M=256][K = ic*9 + fy*3 + fx]
    int m = idx / KTOT, k = idx % KTOT;
    int ic = k / 9, r9 = k % 9, fy = r9 / 3, fx = r9 % 3;
    w1b[idx] = f2bf(w1[(((size_t)m * ICH + ic) * 3 + fy) * 3 + fx]);
    return;
  }
  int i2 = idx - MIDC * KTOT;
  if (i2 < MPAD * 256) {            // w2 -> [M=80 padded][K=256]
    int m = i2 / 256, k = i2 % 256;
    float v = (m < NCLS) ? w2[(size_t)m * 256 + k] : 0.f;
    w2b[i2] = f2bf(v);
  }
}

// ---------------- conv1: implicit-GEMM 3x3 via bf16 WMMA ----------------
// C[m,n] = sum_k A[m,k] * B[k,n],  M=256, N=49152, K=1152
// block = 256 thr (8 waves); block tile 64M x 64N; wave tile 16M x 32N
// LDS tiles are fragment-friendly: A [m][k], B transposed [n][k] so every
// lane's WMMA operand is two aligned b128 LDS loads (ISA 7.12.2 lane layout).
__global__ __launch_bounds__(256) void k_conv1(const __bf16* __restrict__ xb,
                                               const __bf16* __restrict__ w1b,
                                               const float*  __restrict__ b1,
                                               __bf16* __restrict__ y) {
  __shared__ __align__(16) __bf16 As[64 * 32];   // [m][k]
  __shared__ __align__(16) __bf16 Bs[64 * 32];   // [n][k]  (transposed)
  const int tid  = threadIdx.x;
  const int lane = tid & 31;
  const int wave = tid >> 5;
  const int lo   = (lane < 16);
  const int l16  = lane & 15;
  const int wm   = wave & 3;        // 4 M sub-tiles
  const int wn   = wave >> 2;       // 2 N sub-tiles (32 wide each)
  const int mBase = blockIdx.y * 64;
  const int nBase = blockIdx.x * 64;

  v8f acc0 = {}; v8f acc1 = {};

  // im2col coordinates for this thread's B staging row (fixed across K loop)
  const int nl   = tid >> 2;            // local n 0..63
  const int kseg = tid & 3;             // 8-element K segment
  const int nG0  = nBase + nl;
  const int bb   = nG0 / (HH * WW);
  const int hw0  = nG0 - bb * (HH * WW);
  const int h0   = hw0 >> 6, w0 = hw0 & 63;

  for (int kt = 0; kt < KTOT / 32; ++kt) {
    const int k0 = kt * 32;
    __syncthreads();
    {   // A tile: 64x32 [m][k], vectorized 16B global->LDS
      int m = tid >> 2, seg = tid & 3;
      const __bf16* gp = w1b + (size_t)(mBase + m) * KTOT + k0 + seg * 8;
      if (kt + 1 < KTOT / 32)
        __builtin_prefetch(gp + 32, 0, 3);   // global_prefetch_b8, next K-step
      *(uint4*)(As + m * 32 + seg * 8) = *(const uint4*)gp;
    }
    {   // B tile: im2col gather, stored transposed [n][k]; one b128 LDS store
      __bf16 tmp[8];
      #pragma unroll
      for (int j = 0; j < 8; ++j) {
        int k  = k0 + kseg * 8 + j;
        int ic = k / 9, r9 = k % 9, fy = r9 / 3, fx = r9 % 3;
        tmp[j] = xb[(((size_t)(bb * ICH + ic)) * PH + (h0 + fy)) * PW + (w0 + fx)];
      }
      *(uint4*)(Bs + nl * 32 + kseg * 8) = *(const uint4*)tmp;
    }
    __syncthreads();

    // A fragment: 16 contiguous K per lane (k = 16h + s)
    const int arow = (wm * 16 + l16) * 32;
    v16bf a = pack16(*(const uint4*)(As + arow + (lo ? 0 : 16)),
                     *(const uint4*)(As + arow + (lo ? 8 : 24)));

    #pragma unroll
    for (int s = 0; s < 2; ++s) {
      // B fragment: 16 contiguous K per lane in transposed tile
      const int brow = (wn * 32 + s * 16 + l16) * 32 + (lo ? 0 : 16);
      v16bf bf = pack16(*(const uint4*)(Bs + brow),
                        *(const uint4*)(Bs + brow + 8));
      if (s == 0)
        acc0 = __builtin_amdgcn_wmma_f32_16x16x32_bf16(false, a, false, bf,
                                                       (short)0, acc0, false, false);
      else
        acc1 = __builtin_amdgcn_wmma_f32_16x16x32_bf16(false, a, false, bf,
                                                       (short)0, acc1, false, false);
    }
  }

  // epilogue: +bias, relu, store y in [N][K=256] layout (conv2 B-operand ready)
  #pragma unroll
  for (int s = 0; s < 2; ++s) {
    float cf[8];
    __builtin_memcpy(cf, s == 0 ? (void*)&acc0 : (void*)&acc1, sizeof(cf));
    int nG = nBase + wn * 32 + s * 16 + l16;
    #pragma unroll
    for (int i = 0; i < 8; ++i) {
      int mG = mBase + wm * 16 + i + (lo ? 0 : 8);
      float v = cf[i] + b1[mG];
      v = v > 0.f ? v : 0.f;
      y[(size_t)nG * MIDC + mG] = f2bf(v);
    }
  }
}

// ---------------- conv2 (1x1) + softmax + pixel-shuffle, fused ----------------
// block = 256 thr (8 waves); each wave: 16 N columns, all 80 padded M rows.
// B operand comes straight from global y[N][K] as two b128 loads per lane.
__global__ __launch_bounds__(256) void k_conv2(const __bf16* __restrict__ y,
                                               const __bf16* __restrict__ w2b,
                                               const float*  __restrict__ b2,
                                               float* __restrict__ hm) {
  __shared__ __align__(16) unsigned char smem[40960];   // w2 tile -> colbuf reuse
  __bf16* W2s    = (__bf16*)smem;                       // [80][256] bf16
  float*  colbuf = (float*)smem;                        // [128 cols][80]
  const int tid  = threadIdx.x;
  const int lane = tid & 31;
  const int wave = tid >> 5;
  const int lo   = (lane < 16);
  const int l16  = lane & 15;

  {   // stage w2 tile (40KB of the 320KB/WGP LDS)
    const uint4* src = (const uint4*)w2b;
    uint4* dst = (uint4*)W2s;
    for (int i = tid; i < 2560; i += 256) dst[i] = src[i];
  }
  __syncthreads();

  v8f acc[5] = {};
  const int nBase = blockIdx.x * 128 + wave * 16;
  const __bf16* yrow = y + (size_t)(nBase + l16) * MIDC;

  for (int kt = 0; kt < 8; ++kt) {        // K = 256
    const int k0 = kt * 32;
    // B fragment: 32 contiguous bytes of global memory per lane
    const __bf16* yp = yrow + k0 + (lo ? 0 : 16);
    if (kt + 1 < 8) __builtin_prefetch(yp + 32, 0, 3);
    v16bf bf = pack16(*(const uint4*)yp, *(const uint4*)(yp + 8));
    #pragma unroll
    for (int t = 0; t < 5; ++t) {
      const int arow = (t * 16 + l16) * 256 + k0;
      v16bf a = pack16(*(const uint4*)(W2s + arow + (lo ? 0 : 16)),
                       *(const uint4*)(W2s + arow + (lo ? 8 : 24)));
      acc[t] = __builtin_amdgcn_wmma_f32_16x16x32_bf16(false, a, false, bf,
                                                       (short)0, acc[t], false, false);
    }
  }
  __syncthreads();    // everyone done reading W2s; reuse LDS

  #pragma unroll
  for (int t = 0; t < 5; ++t) {
    float cf[8]; __builtin_memcpy(cf, &acc[t], sizeof(cf));
    #pragma unroll
    for (int i = 0; i < 8; ++i) {
      int m = t * 16 + i + (lo ? 0 : 8);
      float bias = (m < NCLS) ? b2[m] : 0.f;
      colbuf[(wave * 16 + l16) * 80 + m] = cf[i] + bias;
    }
  }
  __syncthreads();

  if (tid < 128) {    // one thread per output column: softmax + pixel shuffle
    int w = tid >> 4, n = tid & 15;
    const float* col = colbuf + (w * 16 + n) * 80;
    float mx = col[0];
    for (int c = 1; c < NCLS; ++c) mx = fmaxf(mx, col[c]);
    float sum = 0.f;
    for (int c = 0; c < NCLS; ++c) sum += __expf(col[c] - mx);
    float inv = 1.f / sum;
    int nG = blockIdx.x * 128 + w * 16 + n;
    int b = nG / (HH * WW);
    int rem = nG - b * (HH * WW);
    int h = rem >> 6, xw = rem & 63;
    for (int c = 0; c < 64; ++c) {        // drop dust channel 64
      float v = __expf(col[c] - mx) * inv;
      hm[(size_t)b * HMN + (h * 8 + (c >> 3)) * HMW + xw * 8 + (c & 7)] = v;
    }
  }
}

// ---------------- NMS: greedy-equivalent local max in 9x9 ----------------
__global__ void k_nms(const float* __restrict__ hm, unsigned char* __restrict__ keep) {
  int idx = blockIdx.x * 256 + threadIdx.x;
  if (idx >= BATCH * HMN) return;
  int b = idx / HMN, p = idx % HMN;
  int r = p / HMW, c = p % HMW;
  const float* img = hm + (size_t)b * HMN;
  float v = img[p];
  unsigned char k = 0;
  if (v >= CONF_TH && r > 8 && r <= HMH - 8 && c > 8 && c <= HMW - 8) {
    bool best = true;
    for (int dy = -4; dy <= 4 && best; ++dy) {
      int rr = r + dy;
      for (int dx = -4; dx <= 4; ++dx) {
        if (dy == 0 && dx == 0) continue;
        int cc = c + dx;
        int np = rr * HMW + cc;
        float nv = img[np];
        if (nv < CONF_TH) continue;
        if (nv > v || (nv == v && np < p)) { best = false; break; }
      }
    }
    k = best ? 1 : 0;
  }
  keep[idx] = k;
}

// ---------------- sort: bitonic (working set 33.5MB fits MI455X 192MB L2) ----
__global__ void k_sort_init(const float* __restrict__ hm,
                            float* __restrict__ keys, int* __restrict__ vals) {
  int idx = blockIdx.x * 256 + threadIdx.x;
  if (idx >= BATCH * SORTN) return;
  int b = idx >> 18, i = idx & (SORTN - 1);
  float kk = -__builtin_inff();
  int v = 0;
  if (i < HMN) { kk = hm[(size_t)b * HMN + i]; v = i; }
  keys[idx] = kk; vals[idx] = v;
}

__global__ void k_bitonic(float* __restrict__ keys, int* __restrict__ vals,
                          int j, int k) {
  int tid = blockIdx.x * 256 + threadIdx.x;
  if (tid >= BATCH * SORTN) return;
  int b = tid >> 18, i = tid & (SORTN - 1);
  int ixj = i ^ j;
  if (ixj <= i) return;
  size_t base = (size_t)b << 18;
  float k1 = keys[base + i], k2 = keys[base + ixj];
  int   v1 = vals[base + i], v2 = vals[base + ixj];
  bool firstBetter = (k1 > k2) || (k1 == k2 && v1 < v2);  // conf desc, idx asc
  bool doSwap = ((i & k) == 0) ? !firstBetter : firstBetter;
  if (doSwap) {
    keys[base + i] = k2; keys[base + ixj] = k1;
    vals[base + i] = v2; vals[base + ixj] = v1;
  }
}

// ---------------- gather final outputs (pts, confs, keep) ----------------
__global__ void k_gather(const float* __restrict__ keys, const int* __restrict__ vals,
                         const unsigned char* __restrict__ keep,
                         float* __restrict__ out) {
  int idx = blockIdx.x * 256 + threadIdx.x;
  if (idx >= BATCH * HMN) return;
  int b = idx / HMN, p = idx % HMN;
  size_t sbase = (size_t)b << 18;
  float conf = keys[sbase + p];
  int pix = vals[sbase + p];
  int r = pix / HMW, c = pix % HMW;
  unsigned char kp = keep[(size_t)b * HMN + pix];
  const size_t PTS   = (size_t)BATCH * HMN * 2;
  const size_t CONFS = PTS + (size_t)BATCH * HMN;
  out[((size_t)b * HMN + p) * 2 + 0] = (float)r;
  out[((size_t)b * HMN + p) * 2 + 1] = (float)c;
  out[PTS   + (size_t)b * HMN + p] = conf;
  out[CONFS + (size_t)b * HMN + p] = kp ? 1.f : 0.f;
}

extern "C" void kernel_launch(void* const* d_in, const int* in_sizes, int n_in,
                              void* d_out, int out_size, void* d_ws, size_t ws_size,
                              hipStream_t stream) {
  const float* x  = (const float*)d_in[0];
  const float* w1 = (const float*)d_in[1];
  const float* b1 = (const float*)d_in[2];
  const float* w2 = (const float*)d_in[3];
  const float* b2 = (const float*)d_in[4];
  char* ws = (char*)d_ws;
  __bf16* xb   = (__bf16*)(ws + XB_OFF);
  __bf16* w1b  = (__bf16*)(ws + W1B_OFF);
  __bf16* w2b  = (__bf16*)(ws + W2B_OFF);
  __bf16* y    = (__bf16*)(ws + Y_OFF);
  float*  hm   = (float*)(ws + HM_OFF);
  float*  keys = (float*)(ws + KEY_OFF);
  int*    vals = (int*)(ws + VAL_OFF);
  unsigned char* keep = (unsigned char*)(ws + KEEP_OFF);

  k_prep_x<<<(BATCH * ICH * PH * PW + 255) / 256, 256, 0, stream>>>(x, xb);
  k_prep_w<<<(MIDC * KTOT + MPAD * 256 + 255) / 256, 256, 0, stream>>>(w1, w2, w1b, w2b);

  dim3 g1(NTOT / 64, MIDC / 64);                 // 768 x 4 blocks
  k_conv1<<<g1, 256, 0, stream>>>(xb, w1b, b1, y);

  k_conv2<<<NTOT / 128, 256, 0, stream>>>(y, w2b, b2, hm);

  k_nms<<<(BATCH * HMN + 255) / 256, 256, 0, stream>>>(hm, keep);
  k_sort_init<<<(BATCH * SORTN + 255) / 256, 256, 0, stream>>>(hm, keys, vals);
  for (int k = 2; k <= SORTN; k <<= 1)
    for (int j = k >> 1; j >= 1; j >>= 1)
      k_bitonic<<<(BATCH * SORTN + 255) / 256, 256, 0, stream>>>(keys, vals, j, k);

  k_gather<<<(BATCH * HMN + 255) / 256, 256, 0, stream>>>(keys, vals, keep,
                                                          (float*)d_out);
}
